// LittleMoreEfficientCondConv2d_57750130262555
// MI455X (gfx1250) — matrix operand order repaired and checked by reading, not compiled, not run on hardware
//
#include <hip/hip_runtime.h>
#include <math.h>

// ---------------------------------------------------------------------------
// Problem constants
// ---------------------------------------------------------------------------
#define BATCH   32
#define C_IN    256
#define C_OUT   256
#define HW      56
#define PIX     (HW * HW)          // 3136
#define ATT_IN  512
#define NW      8
#define KTOT    (C_IN * 9)         // 2304

// conv tiling: block = 128 cout x 64 pix, 8 waves, wave = 32 cout x 32 pix
#define PIX_TILE  64               // 3136 / 64 = 49 tiles
#define COUT_TILE 128              // 256 / 128 = 2 tiles
#define K_TILE    32               // cin chunk; 9 * (256/32) = 72 K-steps
#define AS_STR    36               // padded LDS row stride (bf16) = 72 B
#define BS_STR    40               // padded LDS row stride (bf16) = 80 B

typedef __attribute__((ext_vector_type(16))) __bf16 bf16x16;
typedef __attribute__((ext_vector_type(8)))  float  floatx8;
typedef unsigned int uint32x4 __attribute__((ext_vector_type(4)));
typedef int          int32x4  __attribute__((ext_vector_type(4)));
typedef int          int32x8  __attribute__((ext_vector_type(8)));

union FragA { uint2 d[4]; bf16x16 v; };   // 32 bytes
union FragB { uint4 q[2]; bf16x16 v; };   // 32 bytes

// TDM availability (builtin exists on both toolchains; arity differs)
#if defined(__has_builtin)
# if __has_builtin(__builtin_amdgcn_tensor_load_to_lds)
#  define USE_TDM 1
# endif
#endif
#ifndef USE_TDM
# define USE_TDM 0
#endif

__device__ __forceinline__ void wait_tensorcnt0() {
#if defined(__has_builtin) && __has_builtin(__builtin_amdgcn_s_wait_tensorcnt)
    __builtin_amdgcn_s_wait_tensorcnt(0);
#else
    asm volatile("s_wait_tensorcnt 0x0" ::: "memory");
#endif
}

__device__ __forceinline__ unsigned short f2bf(float f) {
    unsigned int u = __float_as_uint(f);
    unsigned int r = u + 0x7FFFu + ((u >> 16) & 1u);   // round-to-nearest-even
    return (unsigned short)(r >> 16);
}

// ---------------------------------------------------------------------------
// Kernel 1: per-(b,cin) mean pool over HxW  +  fp32 -> bf16 conversion of x
// ---------------------------------------------------------------------------
__global__ void pool_convert_kernel(const float* __restrict__ x,
                                    unsigned short* __restrict__ xb,
                                    float* __restrict__ pooled) {
    __shared__ float red[256];
    const int blk = blockIdx.x;            // = b*C_IN + c
    const int tid = threadIdx.x;
    const float* src = x + (size_t)blk * PIX;
    unsigned short* dst = xb + (size_t)blk * PIX;

    float s = 0.f;
    for (int i = tid; i < PIX; i += 256) {
        float v = src[i];
        dst[i] = f2bf(v);
        s += v;
    }
    red[tid] = s;
    __syncthreads();
    for (int off = 128; off > 0; off >>= 1) {
        if (tid < off) red[tid] += red[tid + off];
        __syncthreads();
    }
    if (tid == 0) pooled[blk] = red[0] * (1.0f / (float)PIX);
}

// ---------------------------------------------------------------------------
// Kernel 2: attention MLP + softmax.  grid = B blocks, 256 threads.
// ---------------------------------------------------------------------------
__global__ void attention_kernel(const float* __restrict__ att_x,
                                 const float* __restrict__ pooled,
                                 const float* __restrict__ W1,
                                 const float* __restrict__ b1,
                                 const float* __restrict__ W2,
                                 const float* __restrict__ b2,
                                 const float* __restrict__ Wa,
                                 const float* __restrict__ ba,
                                 float* __restrict__ att) {
    __shared__ float mix_s[C_IN];
    __shared__ float logit_s[NW];
    const int b = blockIdx.x;
    const int c = threadIdx.x;

    float acc = b1[c] + b2[c];
    const float* ax = att_x + (size_t)b * ATT_IN;
    const float* w1 = W1 + (size_t)c * ATT_IN;
    for (int i = 0; i < ATT_IN; ++i) acc += ax[i] * w1[i];
    const float* pl = pooled + (size_t)b * C_IN;
    const float* w2 = W2 + (size_t)c * C_IN;
    for (int i = 0; i < C_IN; ++i) acc += pl[i] * w2[i];
    mix_s[c] = tanhf(acc);
    __syncthreads();

    if (c < NW) {
        float l = ba[c];
        const float* wa = Wa + (size_t)c * C_IN;
        for (int i = 0; i < C_IN; ++i) l += mix_s[i] * wa[i];
        logit_s[c] = l;
    }
    __syncthreads();

    if (c == 0) {
        float m = logit_s[0];
        for (int w = 1; w < NW; ++w) m = fmaxf(m, logit_s[w]);
        float sum = 0.f, e[NW];
        for (int w = 0; w < NW; ++w) { e[w] = expf(logit_s[w] - m); sum += e[w]; }
        float inv = 1.0f / sum;
        for (int w = 0; w < NW; ++w) att[b * NW + w] = e[w] * inv;
    }
}

// ---------------------------------------------------------------------------
// Kernel 3: per-sample conv weights in bf16, *s-major* layout:
//   kb[b][cout][s*256 + cin] = sum_w att[b,w] * W_weight[cout*2304 + cin*9 + s, w]
// ---------------------------------------------------------------------------
__global__ void gen_kernel_kernel(const float* __restrict__ att,
                                  const float* __restrict__ Ww,
                                  unsigned short* __restrict__ kb) {
    __shared__ float attL[BATCH * NW];     // 256 floats
    const int tid = threadIdx.x;
    attL[tid] = att[tid];
    __syncthreads();

    const int idx  = blockIdx.x * 256 + tid;          // < 589824 (dest index)
    const int cout = idx / KTOT;
    const int rem  = idx - cout * KTOT;
    const int s    = rem >> 8;                        // 0..8
    const int cin  = rem & 255;
    const size_t r = (size_t)cout * KTOT + (size_t)cin * 9 + s;  // source row

    const float* w = Ww + r * NW;
    float wv[NW];
    #pragma unroll
    for (int j = 0; j < NW; ++j) wv[j] = w[j];

    for (int b = 0; b < BATCH; ++b) {
        const float* a = &attL[b * NW];
        float acc = 0.f;
        #pragma unroll
        for (int j = 0; j < NW; ++j) acc += a[j] * wv[j];
        kb[(size_t)b * (C_OUT * KTOT) + idx] = f2bf(acc);
    }
}

// ---------------------------------------------------------------------------
// Kernel 4: per-sample bias.  grid = B, block = C_OUT.
// ---------------------------------------------------------------------------
__global__ void gen_bias_kernel(const float* __restrict__ att,
                                const float* __restrict__ Wb,
                                float* __restrict__ biasG) {
    const int b = blockIdx.x;
    const int c = threadIdx.x;
    const float* a = att + b * NW;
    const float* w = Wb + (size_t)c * NW;
    float acc = 0.f;
    #pragma unroll
    for (int j = 0; j < NW; ++j) acc += a[j] * w[j];
    biasG[b * C_OUT + c] = acc;
}

// ---------------------------------------------------------------------------
// Kernel 5: per-sample 3x3 conv as 9 shifted 1x1 GEMMs.
// A tile (128 cout x 32 cin bf16, row stride 4608 B) is DMA'd into LDS by the
// Tensor Data Mover with LDS padding (72 B rows); B tile gathered by VALU;
// compute via v_wmma_f32_16x16x32_bf16 (4 accums per wave).
// grid = (49, 2, 32); block = 256 (8 waves).
// ---------------------------------------------------------------------------
__launch_bounds__(256, 2)
__global__ void condconv_wmma_kernel(const unsigned short* __restrict__ xb,
                                     const unsigned short* __restrict__ kb,
                                     const float* __restrict__ biasG,
                                     float* __restrict__ out) {
    __shared__ __align__(16) unsigned short As[COUT_TILE * AS_STR]; // 9.2 KB
    __shared__ __align__(16) unsigned short Bs[PIX_TILE * BS_STR];  // 5.0 KB
    __shared__ int   xoff9[9 * PIX_TILE];                           // 2.25 KB
    __shared__ float biasT[COUT_TILE];

    const int tid   = threadIdx.x;
    const int ptile = blockIdx.x;           // 0..48
    const int ctile = blockIdx.y;           // 0..1
    const int b     = blockIdx.z;           // 0..31
    const int cout0 = ctile * COUT_TILE;
    const int p0    = ptile * PIX_TILE;

    // one-time tables: per (s, pix) shifted input offset (or -1 for pad)
    for (int i = tid; i < 9 * PIX_TILE; i += 256) {
        int s   = i >> 6;           // 0..8
        int pix = i & 63;
        int p   = p0 + pix;
        int oh  = p / HW;
        int ow  = p - oh * HW;
        int ih  = oh + (s / 3) - 1;
        int iw  = ow + (s % 3) - 1;
        xoff9[i] = ((unsigned)ih < (unsigned)HW && (unsigned)iw < (unsigned)HW)
                     ? (ih * HW + iw) : -1;
    }
    if (tid < COUT_TILE) biasT[tid] = biasG[b * C_OUT + cout0 + tid];

    const int lane = tid & 31;
    const int wid  = tid >> 5;      // 0..7
    const int wy   = wid >> 1;      // 0..3 -> 32-cout group
    const int wx   = wid & 1;       // 0..1 -> 32-pix group
    const int lm   = lane & 15;
    const int lh   = lane >> 4;     // 0/1

    floatx8 c00, c01, c10, c11;
    #pragma unroll
    for (int i = 0; i < 8; ++i) { c00[i] = 0.f; c01[i] = 0.f; c10[i] = 0.f; c11[i] = 0.f; }

    const unsigned short* kbb = kb + (size_t)b * (C_OUT * KTOT);
    const unsigned short* xbb = xb + (size_t)b * (C_IN * PIX);

#if USE_TDM
    // ---- Tensor DMA descriptor (D#) for the A tile, per ISA ch.8 ----
    // group1: wg_mask=0 | data_size=1(2B)<<16 | pad_enable<<20 |
    //         pad_interval=3(64B)<<22 | pad_amount=1(8B)<<25
    // tensor_dim0=2304 [79:48], tensor_dim1=256 [111:80],
    // tile_dim0=32 [127:112], tile_dim1=128 [143:128], dim0_stride=2304 [207:160]
    int32x8 g1;
    g1[0] = (int)((1u << 16) | (1u << 20) | (3u << 22) | (1u << 25));
    g1[1] = (int)((KTOT & 0xFFFFu) << 16);                   // dim0 lo16 @ [63:48]
    g1[2] = (int)(((KTOT >> 16) & 0xFFFFu) | ((C_OUT & 0xFFFFu) << 16));
    g1[3] = (int)(((C_OUT >> 16) & 0xFFFFu) | ((unsigned)K_TILE << 16)); // tile_dim0
    g1[4] = (int)(COUT_TILE);                                // tile_dim1 @ [143:128]
    g1[5] = (int)(KTOT);                                     // dim0_stride lo32
    g1[6] = 0;
    g1[7] = 0;
    const int32x4 gz = {0, 0, 0, 0};
    const unsigned lds_base = (unsigned)(size_t)(&As[0]);    // LDS byte offset
    const unsigned long long gbase =
        (unsigned long long)(size_t)(kbb + (size_t)cout0 * KTOT);
#endif

    for (int s = 0; s < 9; ++s) {
        for (int c0 = 0; c0 < C_IN; c0 += K_TILE) {
            __syncthreads();   // prev-step LDS consumers done; tables ready
            const int scol = s * C_IN + c0;

#if USE_TDM
            // ---- A tile via Tensor Data Mover (one issue from wave 0) ----
            if (wid == 0) {
                unsigned long long ga = gbase + (unsigned long long)scol * 2ull;
                uint32x4 g0;
                g0[0] = 1u;                                   // count=1 (valid)
                g0[1] = lds_base;                             // lds_addr
                g0[2] = (unsigned)ga;                         // global_addr lo
                g0[3] = (unsigned)(ga >> 32) | (2u << 30);    // addr hi | type=2
# if __clang_major__ >= 23
                const int32x8 z8 = {0, 0, 0, 0, 0, 0, 0, 0};
                __builtin_amdgcn_tensor_load_to_lds(g0, g1, gz, gz, z8, 0);
# else
                __builtin_amdgcn_tensor_load_to_lds(g0, g1, gz, gz, 0);
# endif
            }
#else
            // ---- A tile via VALU staging (fallback) ----
            #pragma unroll
            for (int i = 0; i < 16; ++i) {
                int lin = tid + i * 256;
                int r = lin >> 5, c = lin & 31;
                As[r * AS_STR + c] = kbb[(size_t)(cout0 + r) * KTOT + scol + c];
            }
#endif
            // ---- B tile: 64 pix x 32 cin, shifted rows via xoff9 ----
            const int sbase = s * PIX_TILE;
            #pragma unroll
            for (int i = 0; i < 8; ++i) {
                int lin = tid + i * 256;
                int pix = lin & 63, kl = lin >> 6;
                int off = xoff9[sbase + pix];
                int o2  = off < 0 ? 0 : off;
                unsigned short v = xbb[(c0 + kl) * PIX + o2];
                if (off < 0) v = 0;
                Bs[pix * BS_STR + kl] = v;
            }
#if USE_TDM
            if (wid == 0) wait_tensorcnt0();   // A tile landed in LDS
#endif
            __syncthreads();

            // ---- A fragments (16x32 bf16): lane=M, elem e->K=e+16*lh
            FragA fa0, fa1;
            const int m0 = wy * 32 + lm;
            #pragma unroll
            for (int j = 0; j < 4; ++j) {
                int kstart = lh * 16 + j * 4;
                fa0.d[j] = *reinterpret_cast<const uint2*>(&As[m0 * AS_STR + kstart]);
                fa1.d[j] = *reinterpret_cast<const uint2*>(&As[(m0 + 16) * AS_STR + kstart]);
            }
            // ---- B fragments (32x16 bf16): lane=N, elem e -> K = e + 16*lh
            const int kbase = lh * 16;
            const int n0 = wx * 32 + lm;
            FragB fb0, fb1;
            fb0.q[0] = *reinterpret_cast<const uint4*>(&Bs[n0 * BS_STR + kbase]);
            fb0.q[1] = *reinterpret_cast<const uint4*>(&Bs[n0 * BS_STR + kbase + 8]);
            fb1.q[0] = *reinterpret_cast<const uint4*>(&Bs[(n0 + 16) * BS_STR + kbase]);
            fb1.q[1] = *reinterpret_cast<const uint4*>(&Bs[(n0 + 16) * BS_STR + kbase + 8]);

            c00 = __builtin_amdgcn_wmma_f32_16x16x32_bf16(false, fa0.v, false, fb0.v,
                                                          (short)0, c00, false, false);
            c01 = __builtin_amdgcn_wmma_f32_16x16x32_bf16(false, fa0.v, false, fb1.v,
                                                          (short)0, c01, false, false);
            c10 = __builtin_amdgcn_wmma_f32_16x16x32_bf16(false, fa1.v, false, fb0.v,
                                                          (short)0, c10, false, false);
            c11 = __builtin_amdgcn_wmma_f32_16x16x32_bf16(false, fa1.v, false, fb1.v,
                                                          (short)0, c11, false, false);
        }
    }

    // ---- epilogue: C/D layout VGPR r -> M = r + 8*lh, N = lane&15 ----
    #pragma unroll
    for (int r = 0; r < 8; ++r) {
        int m   = lh * 8 + r;
        int ma  = wy * 32 + m;          // rows for fa0
        int mb  = ma + 16;              // rows for fa1
        float bva = biasT[ma];
        float bvb = biasT[mb];
        size_t pa = (size_t)p0 + wx * 32 + lm;
        size_t basea = ((size_t)b * C_OUT + cout0 + ma) * PIX + pa;
        size_t baseb = ((size_t)b * C_OUT + cout0 + mb) * PIX + pa;
        out[basea]      = c00[r] + bva;
        out[basea + 16] = c01[r] + bva;
        out[baseb]      = c10[r] + bvb;
        out[baseb + 16] = c11[r] + bvb;
    }
}

// ---------------------------------------------------------------------------
// Host side
// ---------------------------------------------------------------------------
extern "C" void kernel_launch(void* const* d_in, const int* in_sizes, int n_in,
                              void* d_out, int out_size, void* d_ws, size_t ws_size,
                              hipStream_t stream) {
    (void)in_sizes; (void)n_in; (void)out_size; (void)ws_size;

    const float* x       = (const float*)d_in[0];
    const float* att_x   = (const float*)d_in[1];
    const float* W_att1  = (const float*)d_in[2];
    const float* b_att1  = (const float*)d_in[3];
    const float* W_att2  = (const float*)d_in[4];
    const float* b_att2  = (const float*)d_in[5];
    const float* W_att   = (const float*)d_in[6];
    const float* b_att   = (const float*)d_in[7];
    const float* W_weight= (const float*)d_in[8];
    const float* W_bias  = (const float*)d_in[9];
    float* out = (float*)d_out;

    // workspace layout (total ~89.2 MB)
    char* ws = (char*)d_ws;
    float*          attG   = (float*)(ws + 0);                       //  1 KB
    float*          pooled = (float*)(ws + 1024);                    // 32 KB
    float*          biasG  = (float*)(ws + 33792);                   // 32 KB
    unsigned short* xbf    = (unsigned short*)(ws + 66560);          // 51.4 MB
    unsigned short* kbf    = (unsigned short*)(ws + 51446784ull);    // 37.7 MB

    pool_convert_kernel<<<BATCH * C_IN, 256, 0, stream>>>(x, xbf, pooled);

    attention_kernel<<<BATCH, 256, 0, stream>>>(att_x, pooled,
                                                W_att1, b_att1, W_att2, b_att2,
                                                W_att, b_att, attG);

    gen_kernel_kernel<<<(C_OUT * KTOT) / 256, 256, 0, stream>>>(attG, W_weight, kbf);

    gen_bias_kernel<<<BATCH, C_OUT, 0, stream>>>(attG, W_bias, biasG);

    dim3 grid(PIX / PIX_TILE, C_OUT / COUT_TILE, BATCH);   // (49, 2, 32)
    condconv_wmma_kernel<<<grid, 256, 0, stream>>>(xbf, kbf, biasG, out);
}
